// HeterogeneousHORN_24309514895362
// MI455X (gfx1250) — matrix operand order repaired and checked by reading, not compiled, not run on hardware
//
#include <hip/hip_runtime.h>

#define NODES 512
#define SEQ   784
#define BATCH 256
#define OUTC  10

#define NGROUP 16          // batch groups (16 rows each)
#define NCOL   4           // column groups (128 nodes each) -> 64 blocks total
#define NBLK   (NGROUP * NCOL)

typedef __attribute__((ext_vector_type(16))) _Float16 v16h;
typedef __attribute__((ext_vector_type(8)))  _Float16 v8h;
typedef __attribute__((ext_vector_type(8)))  float    v8f;

// ---- workspace layout (bytes) ----
#define OFF_BAR  0                                   // 16 groups x 128B barrier lines
#define OFF_WHH  2048                                // w_hh as f16: 512*512*2
#define OFF_INT  (OFF_WHH + NODES*NODES*2)           // input transposed (T,B) f32
#define OFF_Y0   (OFF_INT + SEQ*BATCH*4)             // y ping f16
#define OFF_Y1   (OFF_Y0 + BATCH*NODES*2)            // y pong f16
#define OFF_XF   (OFF_Y1 + BATCH*NODES*2)            // x_final f32

// ------------------------------------------------------------------
// Prologue: reset per-group barriers, w_hh -> f16, transpose input
// to (T, B), zero y buffer 0.
// ------------------------------------------------------------------
__global__ void horn_prologue(const float* __restrict__ in_seq,
                              const float* __restrict__ w_hh,
                              unsigned* __restrict__ bar,
                              _Float16* __restrict__ whh_h,
                              float* __restrict__ inT,
                              _Float16* __restrict__ y0) {
  const int stride = gridDim.x * blockDim.x;
  const int tid = blockIdx.x * blockDim.x + threadIdx.x;
  if (tid < 512) bar[tid] = 0u;                       // whole 2KB barrier region
  for (int i = tid; i < NODES * NODES; i += stride)
    whh_h[i] = (_Float16)w_hh[i];
  for (int i = tid; i < SEQ * BATCH; i += stride) {
    const int t = i / BATCH, b = i % BATCH;           // write coalesced
    inT[i] = in_seq[b * SEQ + t];
  }
  for (int i = tid; i < BATCH * NODES; i += stride)
    y0[i] = (_Float16)0.0f;
}

// ------------------------------------------------------------------
// Split group barrier (count + generation), agent scope.
// arrive: release fence + post arrival, return generation to wait on.
// wait:   cheap relaxed spin (thread 0), then one acquire fence.
// ------------------------------------------------------------------
__device__ __forceinline__ unsigned bar_arrive(unsigned* gb, unsigned nblk) {
  unsigned gen = 0;
  __builtin_amdgcn_fence(__ATOMIC_RELEASE, "agent");  // drain stores to L2
  __syncthreads();          // all waves' stores fenced; also orders LDS writes
  if (threadIdx.x == 0) {
    gen = __hip_atomic_load(gb + 1, __ATOMIC_RELAXED, __HIP_MEMORY_SCOPE_AGENT);
    unsigned arrived = __hip_atomic_fetch_add(gb, 1u, __ATOMIC_RELEASE, __HIP_MEMORY_SCOPE_AGENT);
    if (arrived == nblk - 1u) {
      __hip_atomic_store(gb, 0u, __ATOMIC_RELAXED, __HIP_MEMORY_SCOPE_AGENT);
      __hip_atomic_fetch_add(gb + 1, 1u, __ATOMIC_RELEASE, __HIP_MEMORY_SCOPE_AGENT);
    }
  }
  return gen;               // valid in thread 0 (its only consumer)
}

__device__ __forceinline__ void bar_wait(unsigned* gb, unsigned gen) {
  if (threadIdx.x == 0) {
    // relaxed DEV-scope poll: bypasses WGP$ without a global_inv per poll
    while (__hip_atomic_load(gb + 1, __ATOMIC_RELAXED, __HIP_MEMORY_SCOPE_AGENT) == gen)
      __builtin_amdgcn_s_sleep(1);
  }
  __syncthreads();
  __builtin_amdgcn_fence(__ATOMIC_ACQUIRE, "agent"); // global_inv once
}

// Build a 16x32 f16 A fragment from two 16B chunks (ISA layout:
// lane<16: K = k0+{0..7} and k0+{16..23}; lane>=16: +8).
__device__ __forceinline__ v16h make_a(const _Float16* ap) {
  v8h a0 = *(const v8h*)ap;
  v8h a1 = *(const v8h*)(ap + 16);
  return __builtin_shufflevector(a0, a1, 0,1,2,3,4,5,6,7,8,9,10,11,12,13,14,15);
}

// ------------------------------------------------------------------
// Persistent HORN recurrence. 64 blocks x 256 threads.
// Block = (batch group bg of 16 rows, column group cg of 128 nodes).
// Wave w owns node tile nt = cg*128 + w*16 for bg's 16 batch rows.
// B = w_hh.T fragments pinned in VGPRs, K-permuted so slots 0..3 are
// the block's OWN column chunk: those 4 WMMAs read A from LDS and run
// BEFORE the barrier wait, hiding the L2 sync round-trip.
// ------------------------------------------------------------------
__launch_bounds__(256, 1)
__global__ void horn_main(const float* __restrict__ inT,
                          const _Float16* __restrict__ whh_h,
                          _Float16* __restrict__ yb0,
                          _Float16* __restrict__ yb1,
                          float* __restrict__ xf,
                          unsigned* __restrict__ bar,
                          const float* __restrict__ w_ih,
                          const float* __restrict__ b_ih,
                          const float* __restrict__ b_hh,
                          const float* __restrict__ omega,
                          const float* __restrict__ gamma_p,
                          const float* __restrict__ alpha_p) {
  __shared__ _Float16 yl[16 * 128];   // this block's own y tile, row-major

  const int lane = threadIdx.x & 31;
  const int bg   = blockIdx.x >> 2;                     // 0..15 batch group
  const int cg   = blockIdx.x & 3;                      // 0..3 column group
  const int mt   = bg * 16;                             // batch-tile base
  const int wid  = threadIdx.x >> 5;                    // wave in block
  const int nt   = cg * 128 + wid * 16;                 // node-tile base
  const int l16  = lane & 15;
  const int hi   = lane >> 4;

  const int n     = nt + l16;       // lane's node column (C layout: N = lane%16)
  const int nl    = wid * 16 + l16; // column local to this block (0..127)
  const int row_a = mt + l16;       // lane's A row        (A layout: M = lane%16)

  unsigned* gb = bar + bg * 32;     // this group's 128B barrier line

  // per-node ODE parameters
  const float wi  = w_ih[n];
  const float bi  = b_ih[n];
  const float bh  = b_hh[n];
  const float om  = omega[n];
  const float om2 = om * om;
  const float ga2 = 2.0f * gamma_p[n];
  const float al  = alpha_p[n];
  const float inv_sqrt_n = 0.04419417382415922f;   // 1/sqrt(512)

  // B fragments pinned in VGPRs, permuted: slot s <-> K chunk (cg*4+s)&15.
  // Slots 0..3 = this block's own columns (served from LDS each step).
  v16h bfrag[16];
#pragma unroll
  for (int s = 0; s < 16; ++s) {
    const int kk = (cg * 4 + s) & 15;
    bfrag[s] = make_a(whh_h + (size_t)n * NODES + kk * 32 + hi * 8);
  }

  v8f x = {};   // oscillator position, C layout
  v8f y = {};   // oscillator velocity, C layout

  // init local y tile to zero; bar_arrive's syncthreads publishes it
  for (int i = threadIdx.x; i < 16 * 128; i += 256) yl[i] = (_Float16)0.0f;
  unsigned genw = bar_arrive(gb, NCOL);

  for (int t = 0; t < SEQ; ++t) {
    const _Float16* rd = (t & 1) ? yb1 : yb0;   // siblings' y, previous step
    _Float16*       wr = (t & 1) ? yb0 : yb1;   // y for next step

    // s_t for this lane's 8 batch rows (contiguous 32B, independent of barrier)
    const v8f sv = *(const v8f*)(inT + t * BATCH + mt + 8 * hi);

    // ---- phase 1: local K chunk from LDS, before waiting ----
    v8f acc0 = {}, acc1 = {};
#pragma unroll
    for (int s = 0; s < 4; ++s) {
      const _Float16* lp = &yl[l16 * 128 + s * 32 + hi * 8];
      v16h a = make_a(lp);
      if (s & 1)
        acc1 = __builtin_amdgcn_wmma_f32_16x16x32_f16(
            false, a, false, bfrag[s], (short)0, acc1, false, false);
      else
        acc0 = __builtin_amdgcn_wmma_f32_16x16x32_f16(
            false, a, false, bfrag[s], (short)0, acc0, false, false);
    }

    // ---- wait for siblings' y of this step ----
    bar_wait(gb, genw);

    // ---- phase 2: 12 remote K chunks; issue all loads, drain WMMAs ----
    v16h ar[12];
#pragma unroll
    for (int s = 0; s < 12; ++s) {
      const int kk = (cg * 4 + 4 + s) & 15;
      ar[s] = make_a(rd + (size_t)row_a * NODES + kk * 32 + hi * 8);
    }
#pragma unroll
    for (int s = 0; s < 12; ++s) {
      if (s & 1)
        acc1 = __builtin_amdgcn_wmma_f32_16x16x32_f16(
            false, ar[s], false, bfrag[s + 4], (short)0, acc1, false, false);
      else
        acc0 = __builtin_amdgcn_wmma_f32_16x16x32_f16(
            false, ar[s], false, bfrag[s + 4], (short)0, acc0, false, false);
    }
    const v8f acc = acc0 + acc1;

    // ---- oscillator update; C layout: element v -> m = mt + v + 8*hi ----
#pragma unroll
    for (int v = 0; v < 8; ++v) {
      const int m = mt + v + 8 * hi;
      const float A  = (sv[v] * wi + bi + acc[v] + bh) * inv_sqrt_n;
      const float accel = al * tanhf(A) - ga2 * y[v] - om2 * x[v];
      x[v] += y[v];
      y[v] += accel;
      const _Float16 yh = (_Float16)y[v];
      wr[(size_t)m * NODES + n] = yh;           // publish to siblings
      yl[(v + 8 * hi) * 128 + nl] = yh;         // mirror for local phase 1
    }

    // ---- post arrival immediately; next iteration overlaps the sync ----
    genw = bar_arrive(gb, NCOL);
  }

#pragma unroll
  for (int v = 0; v < 8; ++v)
    xf[(size_t)(mt + v + 8 * hi) * NODES + n] = x[v];
}

// ------------------------------------------------------------------
// Readout: out[b][c] = x_final[b,:] . w_ro[c,:] + b_ro[c]
// ------------------------------------------------------------------
__global__ void horn_readout(const float* __restrict__ xf,
                             const float* __restrict__ w_ro,
                             const float* __restrict__ b_ro,
                             float* __restrict__ out) {
  const int c = blockIdx.x;     // 0..9
  const int b = threadIdx.x;    // 0..255
  const float* xr = xf + (size_t)b * NODES;
  const float* wr = w_ro + (size_t)c * NODES;
  float s = b_ro[c];
#pragma unroll 8
  for (int k = 0; k < NODES; ++k) s += xr[k] * wr[k];
  out[b * OUTC + c] = s;
}

// ------------------------------------------------------------------
extern "C" void kernel_launch(void* const* d_in, const int* in_sizes, int n_in,
                              void* d_out, int out_size, void* d_ws, size_t ws_size,
                              hipStream_t stream) {
  const float* in_seq = (const float*)d_in[0];   // (256, 784)
  const float* w_ih   = (const float*)d_in[1];   // (512, 1)
  const float* b_ih   = (const float*)d_in[2];   // (512,)
  const float* w_hh   = (const float*)d_in[3];   // (512, 512)
  const float* b_hh   = (const float*)d_in[4];   // (512,)
  const float* w_ro   = (const float*)d_in[5];   // (10, 512)
  const float* b_ro   = (const float*)d_in[6];   // (10,)
  const float* omega  = (const float*)d_in[7];   // (512,)
  const float* gammap = (const float*)d_in[8];   // (512,)
  const float* alphap = (const float*)d_in[9];   // (512,)
  float* out = (float*)d_out;                    // (256, 10)

  char* ws = (char*)d_ws;
  unsigned* bar   = (unsigned*)(ws + OFF_BAR);
  _Float16* whh_h = (_Float16*)(ws + OFF_WHH);
  float*    inT   = (float*)   (ws + OFF_INT);
  _Float16* yb0   = (_Float16*)(ws + OFF_Y0);
  _Float16* yb1   = (_Float16*)(ws + OFF_Y1);
  float*    xf    = (float*)   (ws + OFF_XF);

  horn_prologue<<<512, 256, 0, stream>>>(in_seq, w_hh, bar, whh_h, inT, yb0);

  horn_main<<<NBLK, 256, 0, stream>>>(inT, whh_h, yb0, yb1, xf, bar,
                                      w_ih, b_ih, b_hh, omega, gammap, alphap);

  horn_readout<<<OUTC, BATCH, 0, stream>>>(xf, w_ro, b_ro, out);
}
